// PPGN_69028714381400
// MI455X (gfx1250) — matrix operand order repaired and checked by reading, not compiled, not run on hardware
//
#include <hip/hip_runtime.h>
#include <hip/hip_bf16.h>

#define BATCH 8
#define NMAT  256
#define PPOS  65536          // NMAT*NMAT
#define EMBC  64
#define NEDGE 32768

typedef __attribute__((ext_vector_type(16))) _Float16 v16h;
typedef __attribute__((ext_vector_type(8)))  _Float16 v8h;
typedef __attribute__((ext_vector_type(8)))  float    v8f;

// ---------------------------------------------------------------- utilities
__global__ void zero_f32_kernel(float* p, size_t n) {
    size_t i = (size_t)blockIdx.x * blockDim.x + threadIdx.x;
    if (i < n) p[i] = 0.0f;
}

__global__ void scatter_adj_kernel(const int* __restrict__ ei,
                                   const int* __restrict__ batch,
                                   float* __restrict__ adj) {
    int e = blockIdx.x * blockDim.x + threadIdx.x;
    if (e >= NEDGE) return;
    int src = ei[e];
    int dst = ei[NEDGE + e];
    int g   = batch[src];
    int lu  = src - g * NMAT;
    int lv  = dst - g * NMAT;
    atomicAdd(&adj[((size_t)g * NMAT + lu) * NMAT + lv], 1.0f);
}

// Pack f32 weight matrix [64, ldW] (cols wcol0..wcol0+cin) into f16 WMMA A-tile
// lane order: out[((oTile*(cin/32)+kb)*32 + lane)*16 + t]
//   lane<16: row=lane,    k=kb*32+t
//   lane>=16: row=lane-16, k=kb*32+16+t (hi*16)
__global__ void pack_w_kernel(const float* __restrict__ W, int ldW, int wcol0, int cin,
                              _Float16* __restrict__ out) {
    int idx = blockIdx.x * blockDim.x + threadIdx.x;
    int total = 4 * (cin / 32) * 32 * 16;
    if (idx >= total) return;
    int t    = idx & 15;
    int lane = (idx >> 4) & 31;
    int kb   = (idx >> 9) % (cin / 32);
    int ot   = (idx >> 9) / (cin / 32);
    int row  = lane & 15, hi = lane >> 4;
    int k    = kb * 32 + hi * 16 + t;
    out[idx] = (_Float16)W[(size_t)(ot * 16 + row) * ldW + wcol0 + k];
}

// rb0 input conv: z has channels (adj, 0) -> out[b][p][o] = relu(w[o,0]*adj + b[o])
__global__ void rb0_inconv_kernel(const float* __restrict__ adj,
                                  const float* __restrict__ w,   // [64,2]
                                  const float* __restrict__ bias,
                                  _Float16* __restrict__ out) {  // [B,P,64]
    __shared__ float sw0[EMBC], sb[EMBC];
    int t = threadIdx.x;
    if (t < EMBC) { sw0[t] = w[t * 2]; sb[t] = bias[t]; }
    __syncthreads();
    size_t pos = (size_t)blockIdx.x * blockDim.x + t;   // over B*P
    float a = adj[pos];
    _Float16* o = out + pos * EMBC;
    #pragma unroll
    for (int c = 0; c < EMBC; c += 8) {
        v8h pk;
        #pragma unroll
        for (int r = 0; r < 8; ++r) {
            float v = fmaf(sw0[c + r], a, sb[c + r]);
            pk[r] = (_Float16)fmaxf(v, 0.0f);
        }
        *(v8h*)(o + c) = pk;
    }
}

// ------------------------------------------------- WMMA channel-mixing conv
// Out[o,p] = sum_c W[o,c] * X[c,p] (+bias, +optional alpha[o]*adj[p])
// WP: pre-packed f16 A-tiles. X layout: position-major [B,P,64]; CIN==128 reads
// second 64 channels from X2.
// OUTMODE 0: pos-major out[b][p][o]  (f16)
// OUTMODE 1: chan-major out[b][o][i*N+k]
// OUTMODE 2: chan-major transposed out[b][o][j*N+i]  (p = i*N+j)
template <int CIN, int OUTMODE, bool RELU, bool ADJADD>
__global__ __launch_bounds__(256)
void conv_wmma_kernel(const _Float16* __restrict__ X,
                      const _Float16* __restrict__ X2,
                      const _Float16* __restrict__ WP,
                      const float* __restrict__ Wsrc, int ldW,   // for ADJADD alpha only
                      const float* __restrict__ bias,
                      const float* __restrict__ adj,
                      _Float16* __restrict__ out) {
    const int lane = threadIdx.x & 31;
    const int wid  = blockIdx.x * (blockDim.x >> 5) + (threadIdx.x >> 5);
    const int PS   = PPOS / 64;                 // 1024 p-strips
    const int total = BATCH * 4 * PS;
    if (wid >= total) return;                   // wave-uniform

    const int b  = wid / (4 * PS);
    const int r0 = wid % (4 * PS);
    const int ot    = r0 / PS;
    const int oBase = ot * 16;
    const int pBase = (r0 % PS) * 64;
    const int row = lane & 15;
    const int hi  = lane >> 4;

    v8f acc[4];
    #pragma unroll
    for (int pb = 0; pb < 4; ++pb)
        #pragma unroll
        for (int r = 0; r < 8; ++r) acc[pb][r] = 0.0f;

    const size_t bbase = (size_t)b * PPOS * 64;

    #pragma unroll
    for (int kb = 0; kb < CIN / 32; ++kb) {
        // A matrix: one lane-coalesced 32-byte load from the packed buffer
        v16h a = *(const v16h*)(WP + ((size_t)(ot * (CIN / 32) + kb) * 32 + lane) * 16);

        const int c0 = kb * 32 + hi * 16;       // 16 consecutive input channels
        const _Float16* xsrc = (CIN == 128 && c0 >= 64)
                                 ? (X2 + bbase + (c0 - 64))
                                 : (X  + bbase + c0);
        #pragma unroll
        for (int pb = 0; pb < 4; ++pb) {
            int p = pBase + pb * 16 + row;
            v16h bm = *(const v16h*)(xsrc + (size_t)p * 64);
            acc[pb] = __builtin_amdgcn_wmma_f32_16x16x32_f16(
                false, a, false, bm, (short)0, acc[pb], false, false);
        }
    }

    const int ob = oBase + hi * 8;
    float bv[8], al[8];
    #pragma unroll
    for (int r = 0; r < 8; ++r) {
        bv[r] = bias[ob + r];
        if (ADJADD) al[r] = Wsrc[(size_t)(ob + r) * ldW];  // sw[o,0] multiplies adj channel
    }

    #pragma unroll
    for (int pb = 0; pb < 4; ++pb) {
        int p = pBase + pb * 16 + row;
        float aj = ADJADD ? adj[(size_t)b * PPOS + p] : 0.0f;
        if (OUTMODE == 0) {
            v8h pk;
            #pragma unroll
            for (int r = 0; r < 8; ++r) {
                float v = acc[pb][r] + bv[r];
                if (ADJADD) v = fmaf(al[r], aj, v);
                if (RELU)   v = fmaxf(v, 0.0f);
                pk[r] = (_Float16)v;
            }
            *(v8h*)(out + ((size_t)b * PPOS + p) * 64 + ob) = pk;
        } else {
            #pragma unroll
            for (int r = 0; r < 8; ++r) {
                float v = acc[pb][r] + bv[r];
                if (ADJADD) v = fmaf(al[r], aj, v);
                if (RELU)   v = fmaxf(v, 0.0f);
                int o = ob + r;
                size_t idx;
                if (OUTMODE == 1) {
                    idx = ((size_t)b * 64 + o) * PPOS + p;
                } else {
                    int i = p >> 8, j = p & 255;
                    idx = ((size_t)b * 64 + o) * PPOS + (size_t)j * NMAT + i;
                }
                out[idx] = (_Float16)v;
            }
        }
    }
}

// ----------------------------------------- per-channel N x N spatial matmul
// Mult[b,c][i,j] = sum_k M1[b,c][i,k] * M2T[b,c][j,k]; output pos-major [B,P,64]
// Each wave: 32(i) x 64(j) output tile (2 A-tiles x 4 B-tiles, 8 accumulators)
__global__ __launch_bounds__(256)
void chan_matmul_kernel(const _Float16* __restrict__ M1,
                        const _Float16* __restrict__ M2T,
                        _Float16* __restrict__ out) {
    const int lane = threadIdx.x & 31;
    const int wid  = blockIdx.x * (blockDim.x >> 5) + (threadIdx.x >> 5);
    if (wid >= BATCH * 64 * 32) return;         // 8 * 64ch * (8 iGroups * 4 jStrips)
    const int b  = wid >> 11;
    const int c  = (wid >> 5) & 63;
    const int ig = (wid >> 2) & 7;
    const int js = wid & 3;
    const int iBase = ig * 32, jBase = js * 64;
    const int row = lane & 15;
    const int hi  = lane >> 4;
    const size_t mbase = ((size_t)b * 64 + c) * PPOS;

    v8f acc[2][4];
    #pragma unroll
    for (int ii = 0; ii < 2; ++ii)
        #pragma unroll
        for (int jb = 0; jb < 4; ++jb)
            #pragma unroll
            for (int r = 0; r < 8; ++r) acc[ii][jb][r] = 0.0f;

    #pragma unroll
    for (int kb = 0; kb < NMAT / 32; ++kb) {
        v16h a[2];
        #pragma unroll
        for (int ii = 0; ii < 2; ++ii) {
            const _Float16* ar = M1 + mbase + (size_t)(iBase + ii * 16 + row) * NMAT
                                 + kb * 32 + hi * 16;
            v8h alo = *(const v8h*)ar;
            v8h ahi = *(const v8h*)(ar + 8);
            #pragma unroll
            for (int t = 0; t < 8; ++t) { a[ii][t] = alo[t]; a[ii][8 + t] = ahi[t]; }
        }
        #pragma unroll
        for (int jb = 0; jb < 4; ++jb) {
            int j = jBase + jb * 16 + row;
            v16h bm = *(const v16h*)(M2T + mbase + (size_t)j * NMAT + kb * 32 + hi * 16);
            acc[0][jb] = __builtin_amdgcn_wmma_f32_16x16x32_f16(
                false, a[0], false, bm, (short)0, acc[0][jb], false, false);
            acc[1][jb] = __builtin_amdgcn_wmma_f32_16x16x32_f16(
                false, a[1], false, bm, (short)0, acc[1][jb], false, false);
        }
    }

    #pragma unroll
    for (int ii = 0; ii < 2; ++ii)
        #pragma unroll
        for (int jb = 0; jb < 4; ++jb) {
            int j = jBase + jb * 16 + row;
            #pragma unroll
            for (int r = 0; r < 8; ++r) {
                int i = iBase + ii * 16 + hi * 8 + r;
                out[((size_t)b * PPOS + (size_t)i * NMAT + j) * 64 + c] =
                    (_Float16)acc[ii][jb][r];
            }
        }
}

// ------------------------------------------------------- pooling + FC head
__global__ void pool_kernel(const _Float16* __restrict__ z, float* __restrict__ h) {
    int bc = blockIdx.x;          // B*64 blocks
    int b = bc >> 6, c = bc & 63;
    int t = threadIdx.x;
    float s = 0.0f, sd = 0.0f;
    for (int p = t; p < PPOS; p += 256) {
        float v = (float)z[((size_t)b * PPOS + p) * 64 + c];
        s += v;
        if ((p >> 8) == (p & 255)) sd += v;
    }
    __shared__ float sh[256], shd[256];
    sh[t] = s; shd[t] = sd;
    __syncthreads();
    for (int off = 128; off > 0; off >>= 1) {
        if (t < off) { sh[t] += sh[t + off]; shd[t] += shd[t + off]; }
        __syncthreads();
    }
    if (t == 0) {
        h[b * 128 + c]      = shd[0] / (float)NMAT;
        h[b * 128 + 64 + c] = (sh[0] - shd[0]) / (float)(NMAT * (NMAT - 1));
    }
}

__global__ void fc_kernel(const float* __restrict__ h,
                          const float* __restrict__ w1, const float* __restrict__ b1,
                          const float* __restrict__ w2, const float* __restrict__ b2,
                          float* __restrict__ out) {
    __shared__ float hid[BATCH][EMBC];
    int t = threadIdx.x;          // 64 threads
    for (int b = 0; b < BATCH; ++b) {
        float s = b1[t];
        for (int k = 0; k < 128; ++k) s = fmaf(w1[t * 128 + k], h[b * 128 + k], s);
        hid[b][t] = fmaxf(s, 0.0f);
    }
    __syncthreads();
    if (t < BATCH) {
        float s = b2[0];
        for (int k = 0; k < EMBC; ++k) s = fmaf(w2[k], hid[t][k], s);
        out[t] = s;
    }
}

// ---------------------------------------------------------------- launcher
extern "C" void kernel_launch(void* const* d_in, const int* in_sizes, int n_in,
                              void* d_out, int out_size, void* d_ws, size_t ws_size,
                              hipStream_t stream) {
    const int*   ei    = (const int*)d_in[0];
    const int*   batch = (const int*)d_in[1];
    const float* rb0_m1w1 = (const float*)d_in[2];
    const float* rb0_m1b1 = (const float*)d_in[3];
    const float* rb0_m1w2 = (const float*)d_in[4];
    const float* rb0_m1b2 = (const float*)d_in[5];
    const float* rb0_m2w1 = (const float*)d_in[6];
    const float* rb0_m2b1 = (const float*)d_in[7];
    const float* rb0_m2w2 = (const float*)d_in[8];
    const float* rb0_m2b2 = (const float*)d_in[9];
    const float* rb0_sw   = (const float*)d_in[10];
    const float* rb0_sb   = (const float*)d_in[11];
    const float* rb_m1w1  = (const float*)d_in[12];
    const float* rb_m1b1  = (const float*)d_in[13];
    const float* rb_m1w2  = (const float*)d_in[14];
    const float* rb_m1b2  = (const float*)d_in[15];
    const float* rb_m2w1  = (const float*)d_in[16];
    const float* rb_m2b1  = (const float*)d_in[17];
    const float* rb_m2w2  = (const float*)d_in[18];
    const float* rb_m2b2  = (const float*)d_in[19];
    const float* rb_sw    = (const float*)d_in[20];
    const float* rb_sb    = (const float*)d_in[21];
    const float* fcw1     = (const float*)d_in[22];
    const float* fcb1     = (const float*)d_in[23];
    const float* fcw2     = (const float*)d_in[24];
    const float* fcb2     = (const float*)d_in[25];

    char* ws = (char*)d_ws;
    float* adj = (float*)ws;                               // B*P f32 = 2 MiB
    size_t adjB = (size_t)BATCH * PPOS * sizeof(float);
    const size_t SZ = (size_t)BATCH * PPOS * 64 * sizeof(_Float16);  // 64 MiB each
    _Float16* bufs[4];
    for (int k = 0; k < 4; ++k) bufs[k] = (_Float16*)(ws + adjB + (size_t)k * SZ);
    float*    hbuf = (float*)(ws + adjB + 4 * SZ);         // B*128 f32 (4 KiB)
    _Float16* wpk  = (_Float16*)(ws + adjB + 4 * SZ + 4096);
    const size_t WSLOT = 8192;                             // halves per slot (16 KiB)
    auto slot = [&](int s) { return wpk + (size_t)s * WSLOT; };

    const int CONV_BLOCKS = (BATCH * 4 * (PPOS / 64)) / 8;   // 4096 (8 waves/block)
    const int MM_BLOCKS   = (BATCH * 64 * 32) / 8;           // 2048

    // ---------------------------------------------- weight packing (tiny)
    // slots: 0 rb0_m1w2, 1 rb0_m2w2, 2 rb0_sw(K=64,col0=2), 3+i*5.. per block
    pack_w_kernel<<<16, 256, 0, stream>>>(rb0_m1w2, 64, 0, 64, slot(0));
    pack_w_kernel<<<16, 256, 0, stream>>>(rb0_m2w2, 64, 0, 64, slot(1));
    pack_w_kernel<<<16, 256, 0, stream>>>(rb0_sw,   66, 2, 64, slot(2));
    for (int i = 0; i < 3; ++i) {
        pack_w_kernel<<<16, 256, 0, stream>>>(rb_m1w1 + (size_t)i * 4096, 64, 0, 64, slot(3 + i * 5));
        pack_w_kernel<<<16, 256, 0, stream>>>(rb_m1w2 + (size_t)i * 4096, 64, 0, 64, slot(4 + i * 5));
        pack_w_kernel<<<16, 256, 0, stream>>>(rb_m2w1 + (size_t)i * 4096, 64, 0, 64, slot(5 + i * 5));
        pack_w_kernel<<<16, 256, 0, stream>>>(rb_m2w2 + (size_t)i * 4096, 64, 0, 64, slot(6 + i * 5));
        pack_w_kernel<<<32, 256, 0, stream>>>(rb_sw  + (size_t)i * 8192, 128, 0, 128, slot(7 + i * 5));
    }

    // adjacency
    zero_f32_kernel<<<(BATCH * PPOS) / 256, 256, 0, stream>>>(adj, (size_t)BATCH * PPOS);
    scatter_adj_kernel<<<NEDGE / 256, 256, 0, stream>>>(ei, batch, adj);

    _Float16 *t = bufs[0], *m1 = bufs[1], *m2 = bufs[2], *z = bufs[3];

    // ------------------------------------------------------------- rb0
    rb0_inconv_kernel<<<(BATCH * PPOS) / 256, 256, 0, stream>>>(adj, rb0_m1w1, rb0_m1b1, t);
    conv_wmma_kernel<64, 1, true, false><<<CONV_BLOCKS, 256, 0, stream>>>(
        t, nullptr, slot(0), nullptr, 0, rb0_m1b2, nullptr, m1);
    rb0_inconv_kernel<<<(BATCH * PPOS) / 256, 256, 0, stream>>>(adj, rb0_m2w1, rb0_m2b1, t);
    conv_wmma_kernel<64, 2, true, false><<<CONV_BLOCKS, 256, 0, stream>>>(
        t, nullptr, slot(1), nullptr, 0, rb0_m2b2, nullptr, m2);
    chan_matmul_kernel<<<MM_BLOCKS, 256, 0, stream>>>(m1, m2, t);     // mult -> t
    // skip: sum_c sw[o,2+c]*mult[c] + sw[o,0]*adj + sb   (channel 1 of z is zero)
    conv_wmma_kernel<64, 0, false, true><<<CONV_BLOCKS, 256, 0, stream>>>(
        t, nullptr, slot(2), rb0_sw, 66, rb0_sb, adj, z);

    // ------------------------------------------------- regular blocks 1..3
    _Float16* fb[3] = { t, m1, m2 };   // free buffers; z holds current state
    for (int i = 0; i < 3; ++i) {
        const float* b11 = rb_m1b1 + (size_t)i * 64;
        const float* b12 = rb_m1b2 + (size_t)i * 64;
        const float* b21 = rb_m2b1 + (size_t)i * 64;
        const float* b22 = rb_m2b2 + (size_t)i * 64;
        const float* sbi = rb_sb   + (size_t)i * 64;
        _Float16 *tt = fb[0], *mm1 = fb[1], *mm2 = fb[2];

        conv_wmma_kernel<64, 0, true, false><<<CONV_BLOCKS, 256, 0, stream>>>(
            z, nullptr, slot(3 + i * 5), nullptr, 0, b11, nullptr, tt);
        conv_wmma_kernel<64, 1, true, false><<<CONV_BLOCKS, 256, 0, stream>>>(
            tt, nullptr, slot(4 + i * 5), nullptr, 0, b12, nullptr, mm1);
        conv_wmma_kernel<64, 0, true, false><<<CONV_BLOCKS, 256, 0, stream>>>(
            z, nullptr, slot(5 + i * 5), nullptr, 0, b21, nullptr, tt);
        conv_wmma_kernel<64, 2, true, false><<<CONV_BLOCKS, 256, 0, stream>>>(
            tt, nullptr, slot(6 + i * 5), nullptr, 0, b22, nullptr, mm2);
        chan_matmul_kernel<<<MM_BLOCKS, 256, 0, stream>>>(mm1, mm2, tt);   // mult -> tt
        conv_wmma_kernel<128, 0, false, false><<<CONV_BLOCKS, 256, 0, stream>>>(
            z, tt, slot(7 + i * 5), nullptr, 0, sbi, nullptr, mm1);        // z_new -> mm1

        _Float16* oldz = z;
        z = mm1;
        fb[1] = oldz;          // fb = { tt, oldz, mm2 }
    }

    // --------------------------------------------------------- pool + head
    pool_kernel<<<BATCH * 64, 256, 0, stream>>>(z, hbuf);
    fc_kernel<<<1, 64, 0, stream>>>(hbuf, fcw1, fcb1, fcw2, fcb2, (float*)d_out);
}
